// CeNNLayer_38036230373875
// MI455X (gfx1250) — compile-verified, matrix operand order checked by reading, not run on hardware
//
#include <hip/hip_runtime.h>

// ---------------- CDNA5 (gfx1250) CeNN layer via bf16 WMMA implicit GEMM ----
// 12 convs x 19.3 GFLOP = 232 GFLOP; working set ~= 192MB = L2 size. Math:
// v_wmma_f32_16x16x32_bf16 (f32 accumulate), fragments fed by ds_load_b128,
// weights staged to LDS with global_load_async_to_lds_b128 (ASYNCcnt).
// nonlin+bf16 activation production is fused into the conv epilogue with
// ping-pong activation buffers (removes 10 elementwise passes ~1GB traffic).

#define DTSTEP 0.1f
#define ALPHA 0.01f

typedef __attribute__((ext_vector_type(16))) __bf16 v16bf;
typedef __attribute__((ext_vector_type(8)))  float  v8f;

static constexpr int NB = 16, C = 64, H = 128, W = 128;
static constexpr long P  = (long)NB * H * W;       // 262144 pixels
static constexpr long PC = P * C;                  // 16,777,216 elements
static constexpr int ACT_COLS  = W + 2;            // halo'd row width
static constexpr int ACT_ELEMS = 3 * ACT_COLS * C; // 24960 bf16 in LDS
static constexpr int WT_ELEMS  = 9 * C * C;        // 36864 bf16 in LDS
static constexpr size_t SMEM_BYTES = (size_t)(ACT_ELEMS + WT_ELEMS) * 2; // 123,648 B

__device__ __forceinline__ unsigned short f2bf(float f) {
  unsigned int u = __float_as_uint(f);
  u += 0x7FFFu + ((u >> 16) & 1u);          // round-to-nearest-even
  return (unsigned short)(u >> 16);
}
__device__ __forceinline__ float nonlin(float x) {
  float y = fminf(x, 1.0f + ALPHA * (x - 1.0f));
  return fmaxf(y, -1.0f + ALPHA * (y + 1.0f));
}

// x: NCHW f32 -> act: NHWC bf16 (channels contiguous = GEMM K dim)
__global__ __launch_bounds__(256) void k_x_to_nhwc(const float* __restrict__ x,
                                                   unsigned short* __restrict__ act) {
  long i = (long)blockIdx.x * blockDim.x + threadIdx.x;
  if (i >= PC) return;
  int w = (int)(i & 127), h = (int)((i >> 7) & 127);
  int c = (int)((i >> 14) & 63); int n = (int)(i >> 20);
  long o = (((long)n * H + h) * W + w) * C + c;
  act[o] = f2bf(x[i]);
}

// w: [Co][Ci][3][3] f32 -> wt: [co][ky*3+kx][ci] bf16 (ci contiguous; small
// per-kidx constants so DS immediates stay tiny in the unrolled conv loop)
__global__ __launch_bounds__(256) void k_prep_w(const float* __restrict__ w,
                                                unsigned short* __restrict__ wt) {
  int i = blockIdx.x * blockDim.x + threadIdx.x;
  if (i >= WT_ELEMS) return;
  int ci = i & 63;
  int t  = i >> 6;            // co*9 + kidx
  int kidx = t % 9;
  int co   = t / 9;
  int ky = kidx / 3, kx = kidx - ky * 3;
  wt[i] = f2bf(w[((co * 64 + ci) * 3 + ky) * 3 + kx]);
}

// out (NCHW f32) <- nonlin(state NHWC)
__global__ __launch_bounds__(256) void k_final(const float* __restrict__ state,
                                               float* __restrict__ out) {
  long i = (long)blockIdx.x * blockDim.x + threadIdx.x;
  if (i >= PC) return;
  int w = (int)(i & 127), h = (int)((i >> 7) & 127);
  int c = (int)((i >> 14) & 63); int n = (int)(i >> 20);
  long s = (((long)n * H + h) * W + w) * C + c;
  out[i] = nonlin(state[s]);
}

// Implicit-GEMM 3x3 conv. One WG per (n,h) row; wave computes 16 pixels x 64 co.
// mode 0: out = acc + bias;                  actOut = bf16(nonlin(out))
// mode 1: out = acc + bias + Z               (input coupling; no act write)
// mode 2: out = (1-dt)*out + dt*(acc+b+cpl); actOut = bf16(nonlin(out))
__global__ __launch_bounds__(256) void k_conv_wmma(
    const unsigned short* __restrict__ act,   // NHWC bf16 (input activations)
    const unsigned short* __restrict__ wt,    // [co][9][ci] bf16
    const float* __restrict__ bias,
    const float* __restrict__ Zc,
    const float* __restrict__ coupling,
    float* __restrict__ out,
    unsigned short* __restrict__ actOut,      // next-iteration activations
    int mode) {
  extern __shared__ unsigned short smem[];
  unsigned short* actS = smem;              // 24960 bf16 (halo'd act slab)
  unsigned short* wS   = smem + ACT_ELEMS;  // 36864 bf16 ([co][kidx][ci])

  const int tid = threadIdx.x;
  const int bh  = blockIdx.x;               // n*H + h
  const int n   = bh / H;
  const int h   = bh - n * H;

  // ---- stage weights: 72KB contiguous copy via async global->LDS (ASYNCcnt)
  {
    const unsigned ldsWBase = (unsigned)(ACT_ELEMS * 2); // byte offset of wS
    for (int i = tid; i < WT_ELEMS / 8; i += 256) {      // 4608 x b128
      unsigned ldsOff = ldsWBase + (unsigned)(i * 16);
      unsigned gOff   = (unsigned)(i * 16);
      asm volatile("global_load_async_to_lds_b128 %0, %1, %2"
                   :: "v"(ldsOff), "v"(gOff), "s"(wt)
                   : "memory");
    }
  }
  // ---- stage 3 halo'd input rows x 130 cols x 64 ch (b128, zero-filled edges)
  {
    for (int t = tid; t < ACT_ELEMS / 8; t += 256) {     // 3120 x b128
      int e   = t * 8;
      int r   = e / (ACT_COLS * C);
      int rem = e - r * (ACT_COLS * C);
      int col = rem >> 6;
      int ci  = rem & 63;                                 // multiple of 8
      int gh  = h - 1 + r;
      int gw  = col - 1;
      uint4 v = {0u, 0u, 0u, 0u};
      if ((unsigned)gh < (unsigned)H && (unsigned)gw < (unsigned)W) {
        long gi = (((long)n * H + gh) * W + gw) * C + ci;
        v = *(const uint4*)(act + gi);
      }
      *(uint4*)(actS + e) = v;
    }
  }
  asm volatile("s_wait_asynccnt 0x0" ::: "memory");
  __syncthreads();

  const int lane  = tid & 31;
  const int wave  = tid >> 5;          // 8 wave32s; wave*16 pixels each => 128 = W
  const int wbase = wave << 4;
  const int half  = lane >> 4;
  const int mrow  = lane & 15;         // A row (pixel) / B column (co) index

  // ISA 16-bit fragment layout (CDNA5 ISA): lane's 8 VGPRs hold two contiguous
  // 16-byte runs: K = [half*8 .. half*8+7] and the same +16 -> 2x ds_load_b128.
  union Frag { uint4 q[2]; v16bf v; };
  v8f acc[4] = {};

  const unsigned short* aBase  = actS + ((wbase + mrow) << 6) + (half << 3);
  const unsigned short* bBase0 = wS + (mrow * 576) + (half << 3);

#pragma unroll
  for (int kb = 0; kb < 18; ++kb) {    // K = 9 taps * 64 ch, 32 per step
    const int kidx = kb >> 1;
    const int ciB  = (kb & 1) << 5;
    const int ky   = kidx / 3;
    const int kx   = kidx - ky * 3;

    Frag a;
    const unsigned short* ap = aBase + ((ky * ACT_COLS + kx) << 6) + ciB;
    a.q[0] = *(const uint4*)(ap);
    a.q[1] = *(const uint4*)(ap + 16);

#pragma unroll
    for (int nt = 0; nt < 4; ++nt) {   // 4 x 16 output channels
      Frag b;
      const unsigned short* bp = bBase0 + nt * (16 * 576) + (kidx << 6) + ciB;
      b.q[0] = *(const uint4*)(bp);
      b.q[1] = *(const uint4*)(bp + 16);
      acc[nt] = __builtin_amdgcn_wmma_f32_16x16x32_bf16(
          false, a.v, false, b.v, (short)0, acc[nt], false, false);
    }
  }

  // Epilogue per the 16x16 f32 C/D layout: lane<16 -> M=r, lane>=16 -> M=r+8
  const long rowBase = (long)bh * W;
#pragma unroll
  for (int nt = 0; nt < 4; ++nt) {
    const int co  = (nt << 4) + (lane & 15);
    const float bv = bias[co];
    const float zv = (mode == 1) ? Zc[co] : 0.0f;
#pragma unroll
    for (int r = 0; r < 8; ++r) {
      const int  m   = r + (half << 3);
      const long idx = (rowBase + wbase + m) * C + co;
      float v = acc[nt][r] + bv + zv;
      if (mode == 2) {
        float s = out[idx];
        v = s + DTSTEP * (-s + v + coupling[idx]);
      }
      out[idx] = v;
      if (mode != 1) actOut[idx] = f2bf(nonlin(v)); // next-iter activations
    }
  }
}

extern "C" void kernel_launch(void* const* d_in, const int* in_sizes, int n_in,
                              void* d_out, int out_size, void* d_ws, size_t ws_size,
                              hipStream_t stream) {
  (void)in_sizes; (void)n_in; (void)out_size; (void)ws_size;
  const float* x  = (const float*)d_in[0];
  const float* rw = (const float*)d_in[1];
  const float* rb = (const float*)d_in[2];
  const float* Aw = (const float*)d_in[3];
  const float* Ab = (const float*)d_in[4];
  const float* Bw = (const float*)d_in[5];
  const float* Bb = (const float*)d_in[6];
  const float* Z  = (const float*)d_in[7];

  char* ws = (char*)d_ws;
  unsigned short* act0 = (unsigned short*)ws;                            // 32 MiB
  unsigned short* act1 = (unsigned short*)(ws + (size_t)PC * 2);         // 32 MiB
  float* state         = (float*)(ws + (size_t)PC * 4);                  // 64 MiB
  float* coup          = (float*)(ws + (size_t)PC * 4 + (size_t)PC * 4); // 64 MiB
  unsigned short* wtR  = (unsigned short*)(ws + (size_t)PC * 4 + (size_t)PC * 8);
  unsigned short* wtA  = wtR + WT_ELEMS;
  unsigned short* wtB  = wtA + WT_ELEMS;

  hipFuncSetAttribute(reinterpret_cast<const void*>(k_conv_wmma),
                      hipFuncAttributeMaxDynamicSharedMemorySize, (int)SMEM_BYTES);

  dim3 blk(256);
  k_x_to_nhwc<<<(int)(PC / 256), blk, 0, stream>>>(x, act0);
  k_prep_w<<<WT_ELEMS / 256, blk, 0, stream>>>(rw, wtR);
  k_prep_w<<<WT_ELEMS / 256, blk, 0, stream>>>(Aw, wtA);
  k_prep_w<<<WT_ELEMS / 256, blk, 0, stream>>>(Bw, wtB);

  // input_coupling = conv(x, B_w) + B_b + Z         (no act production)
  k_conv_wmma<<<NB * H, blk, SMEM_BYTES, stream>>>(act0, wtB, Bb, Z, coup,
                                                   coup, act1, 1);
  // state = conv(x, rescale_w) + rescale_b; act1 = bf16(nonlin(state))
  k_conv_wmma<<<NB * H, blk, SMEM_BYTES, stream>>>(act0, wtR, rb, Z, coup,
                                                   state, act1, 0);

  // 10 Euler steps; act ping-pongs (act0 is dead as x-act after the init convs)
  unsigned short* cur = act1;
  unsigned short* nxt = act0;
  for (int it = 0; it < 10; ++it) {
    k_conv_wmma<<<NB * H, blk, SMEM_BYTES, stream>>>(cur, wtA, Ab, Z, coup,
                                                     state, nxt, 2);
    unsigned short* tmp = cur; cur = nxt; nxt = tmp;
  }
  k_final<<<(int)(PC / 256), blk, 0, stream>>>(state, (float*)d_out);
}
